// Generator_fuspts_13108240187431
// MI455X (gfx1250) — matrix operand order repaired and checked by reading, not compiled, not run on hardware
//
#include <hip/hip_runtime.h>
#include <hip/hip_bf16.h>

// ---------------------------------------------------------------------------
// CDNA5 (gfx1250) PointNet++-MSG + image fusion forward pass.
// GEMM core: v_wmma_f32_16x16x32_f16, fp32->f16 conversion while staging
// through LDS. wave32, 128-thread blocks (4 waves), one 16x16 tile per wave,
// K chunked by 64 (2 WMMAs per LDS stage).  All callers guarantee M % 64 == 0,
// so there are NO row guards; K/N tails use clamped-index + select (no EXEC
// divergence).  Staging uses per-lane base pointers with compile-time row
// offsets so the K loop strength-reduces (no per-chunk 64-bit multiplies).
// ---------------------------------------------------------------------------

typedef __attribute__((ext_vector_type(16))) _Float16 v16h;
typedef __attribute__((ext_vector_type(8)))  float    v8f;

#define BN_EPS 1e-5f

// ----------------------------- WMMA GEMM -----------------------------------
// Y[M,N] = A[M,K] * W[K,N] (+ bias[N]).  Requires M % 64 == 0.
__global__ __launch_bounds__(128)
void gemm_bias_wmma(const float* __restrict__ A, const float* __restrict__ B,
                    const float* __restrict__ bias, float* __restrict__ Y,
                    int M, int K, int N)
{
    __shared__ _Float16 lA[4][16][72];   // per-wave 16x64 A chunk (144B rows)
    __shared__ _Float16 lB[64][24];      // shared 64x16 B chunk (48B rows)

    const int tid  = threadIdx.x;
    const int lane = tid & 31;
    const int wave = tid >> 5;

    const int row0 = (blockIdx.x * 4 + wave) * 16;   // M base of this wave
    const int col0 = blockIdx.y * 16;                // N base of this block

    const int arow = lane & 15;
    const int hi   = lane >> 4;          // A half-wave K interleave selector

    // loop-invariant staging bases
    const float* aBase = A + (long)row0 * K + lane;  // lane = column within chunk
    const int  brow = tid >> 4;                      // 0..7  (B row group)
    const int  bcol = tid & 15;                      // B column within tile
    const int  gcB  = col0 + bcol;
    const int  gcBc = (gcB < N) ? gcB : (N - 1);     // clamped col, hoisted
    const bool cvB  = (gcB < N);

    v8f acc = {};

    for (int kk = 0; kk < K; kk += 64) {
        if (kk + 64 <= K) {
            // -------- interior: unguarded, strength-reduced --------
            #pragma unroll
            for (int r = 0; r < 16; ++r) {
                const float* ap = aBase + (long)r * K + kk;   // r*K is constant
                lA[wave][r][lane]      = (_Float16)ap[0];
                lA[wave][r][lane + 32] = (_Float16)ap[32];
            }
            {
                const float* bp = B + (long)(kk + brow) * N + gcBc;
                #pragma unroll
                for (int e = 0; e < 8; ++e) {
                    float v = bp[(long)(e * 8) * N];          // constant multiple
                    lB[e * 8 + brow][bcol] = (_Float16)(cvB ? v : 0.0f);
                }
            }
        } else {
            // -------- K tail: clamp + select, unconditional loads --------
            #pragma unroll
            for (int r = 0; r < 16; ++r) {
                #pragma unroll
                for (int h = 0; h < 2; ++h) {
                    int gk  = kk + lane + 32 * h;
                    int gkc = (gk < K) ? gk : (K - 1);
                    float v = A[(long)(row0 + r) * K + gkc];
                    lA[wave][r][lane + 32 * h] =
                        (_Float16)((gk < K) ? v : 0.0f);
                }
            }
            #pragma unroll
            for (int e = 0; e < 8; ++e) {
                int gr  = kk + e * 8 + brow;
                int grc = (gr < K) ? gr : (K - 1);
                float v = B[(long)grc * N + gcBc];
                lB[e * 8 + brow][bcol] =
                    (_Float16)((gr < K && cvB) ? v : 0.0f);
            }
        }
        // prefetch next A chunk (global_prefetch_b8, speculative)
        if (kk + 64 < K)
            __builtin_prefetch(&A[(long)(row0 + arow) * K + kk + 64], 0, 1);
        __syncthreads();

        // ---- two 16x16x32 WMMAs over the 64-wide chunk ----
        #pragma unroll
        for (int h = 0; h < 2; ++h) {
            v16h af, bf;
            // A frag: lanes 0-15 -> K {0..7,16..23}, lanes 16-31 -> {8..15,24..31}
            // (two contiguous 8-half runs per lane -> ds_load_b128 pairs)
            #pragma unroll
            for (int e = 0; e < 8; ++e) {
                af[e]     = lA[wave][arow][h * 32 + (hi ? 8 : 0) + e];
                af[e + 8] = lA[wave][arow][h * 32 + (hi ? 24 : 16) + e];
            }
            #pragma unroll
            for (int e = 0; e < 8; ++e) {
                bf[e]     = lB[h * 32 + (hi ? 8 : 0) + e][arow];   // lane = N col, elem = K (same order as A)
                bf[e + 8] = lB[h * 32 + (hi ? 24 : 16) + e][arow];
            }
            acc = __builtin_amdgcn_wmma_f32_16x16x32_f16(
                      false, af, false, bf, (short)0, acc, false, false);
        }
        __syncthreads();
    }

    // ---- epilogue: D layout (vgpr v, lanes 0-15 -> M=v; lanes 16-31 -> M=v+8)
    const int n = col0 + (lane & 15);
    const int mo = (lane < 16) ? 0 : 8;
    if (n < N) {
        float bv = bias ? bias[n] : 0.0f;
        #pragma unroll
        for (int v = 0; v < 8; ++v) {
            int m = row0 + mo + v;
            Y[(long)m * N + n] = acc[v] + bv;
        }
    }
}

// ----------------------- batchnorm statistics ------------------------------
__global__ __launch_bounds__(256)
void colstats(const float* __restrict__ Y, int M, int N,
              float* __restrict__ mean, float* __restrict__ var)
{
    const int c = blockIdx.x;
    const int tid = threadIdx.x;
    float s = 0.0f, s2 = 0.0f;
    for (int i = tid; i < M; i += 256) {
        float v = Y[(long)i * N + c];
        s += v; s2 += v * v;
    }
    __shared__ float sh1[256], sh2[256];
    sh1[tid] = s; sh2[tid] = s2;
    __syncthreads();
    for (int off = 128; off > 0; off >>= 1) {
        if (tid < off) { sh1[tid] += sh1[tid + off]; sh2[tid] += sh2[tid + off]; }
        __syncthreads();
    }
    if (tid == 0) {
        float m = sh1[0] / (float)M;
        mean[c] = m;
        var[c]  = sh2[0] / (float)M - m * m;
    }
}

__global__ void bn_act(float* __restrict__ Y, long total, int N,
                       const float* __restrict__ mean, const float* __restrict__ var,
                       const float* __restrict__ g, const float* __restrict__ b,
                       int relu)
{
    long t = blockIdx.x * (long)blockDim.x + threadIdx.x;
    if (t >= total) return;
    int c = (int)(t % N);
    float v = (Y[t] - mean[c]) * rsqrtf(var[c] + BN_EPS) * g[c] + b[c];
    Y[t] = relu ? fmaxf(v, 0.0f) : v;
}

// --------------------------- FPS (single block) ----------------------------
__global__ __launch_bounds__(1024)
void fps_kernel(const float* __restrict__ xyz, int N, int npoint,
                int* __restrict__ idx, float* __restrict__ dists)
{
    __shared__ float smax[1024];
    __shared__ int   simax[1024];
    __shared__ int   s_last;
    const int tid = threadIdx.x;
    for (int i = tid; i < N; i += 1024) dists[i] = 1e10f;
    if (tid == 0) { idx[0] = 0; s_last = 0; }
    __syncthreads();
    for (int j = 1; j < npoint; ++j) {
        int last = s_last;
        float px = xyz[last * 3 + 0], py = xyz[last * 3 + 1], pz = xyz[last * 3 + 2];
        float best = -1.0f; int bi = 0;
        for (int i = tid; i < N; i += 1024) {
            float dx = xyz[i * 3 + 0] - px;
            float dy = xyz[i * 3 + 1] - py;
            float dz = xyz[i * 3 + 2] - pz;
            float d = dx * dx + dy * dy + dz * dz;
            float dm = fminf(dists[i], d);
            dists[i] = dm;
            if (dm > best) { best = dm; bi = i; }
        }
        smax[tid] = best; simax[tid] = bi;
        __syncthreads();
        for (int off = 512; off > 0; off >>= 1) {
            if (tid < off) {
                bool take = (smax[tid + off] > smax[tid]) ||
                            (smax[tid + off] == smax[tid] && simax[tid + off] < simax[tid]);
                if (take) { smax[tid] = smax[tid + off]; simax[tid] = simax[tid + off]; }
            }
            __syncthreads();
        }
        if (tid == 0) { idx[j] = simax[0]; s_last = simax[0]; }
        __syncthreads();
    }
}

__global__ void gather_xyz(const float* __restrict__ xyz, const int* __restrict__ idx,
                           int S, float* __restrict__ out)
{
    int t = blockIdx.x * blockDim.x + threadIdx.x;
    if (t < S * 3) {
        int s = t / 3, d = t % 3;
        out[t] = xyz[idx[s] * 3 + d];
    }
}

// ------------------------------ ball query ---------------------------------
__global__ void ball_query_kernel(const float* __restrict__ new_xyz,
                                  const float* __restrict__ xyz,
                                  int S, int N, float r2, int ns,
                                  int* __restrict__ gi)
{
    int s = blockIdx.x * blockDim.x + threadIdx.x;
    if (s >= S) return;
    float px = new_xyz[s * 3 + 0], py = new_xyz[s * 3 + 1], pz = new_xyz[s * 3 + 2];
    int cnt = 0, first = 0; bool any = false;
    for (int i = 0; i < N && cnt < ns; ++i) {
        float dx = xyz[i * 3 + 0] - px;
        float dy = xyz[i * 3 + 1] - py;
        float dz = xyz[i * 3 + 2] - pz;
        float d = dx * dx + dy * dy + dz * dz;
        if (d < r2) {
            if (!any) { any = true; first = i; }
            gi[s * ns + cnt] = i; ++cnt;
        }
    }
    if (!any) first = 0;
    for (; cnt < ns; ++cnt) gi[s * ns + cnt] = first;
}

// group rows: [S*ns, 3+Cf]  (centered xyz ++ gathered feats)
__global__ void group_gather(const float* __restrict__ xyz, const float* __restrict__ feats,
                             const float* __restrict__ new_xyz, const int* __restrict__ gi,
                             int S, int ns, int Cf, float* __restrict__ A)
{
    long t = blockIdx.x * (long)blockDim.x + threadIdx.x;
    int Ktot = 3 + Cf;
    long total = (long)S * ns * Ktot;
    if (t >= total) return;
    int c = (int)(t % Ktot);
    long rowt = t / Ktot;
    int j = (int)(rowt % ns);
    int s = (int)(rowt / ns);
    int g = gi[s * ns + j];
    float v;
    if (c < 3) v = xyz[g * 3 + c] - new_xyz[s * 3 + c];
    else       v = feats[(long)g * Cf + (c - 3)];
    A[t] = v;
}

__global__ void maxpool(const float* __restrict__ Y, int S, int ns, int C,
                        float* __restrict__ out, int outStride, int outOffset)
{
    long t = blockIdx.x * (long)blockDim.x + threadIdx.x;
    if (t >= (long)S * C) return;
    int s = (int)(t / C), c = (int)(t % C);
    float m = -3.4e38f;
    for (int j = 0; j < ns; ++j)
        m = fmaxf(m, Y[((long)s * ns + j) * C + c]);
    out[(long)s * outStride + outOffset + c] = m;
}

// --------------------------- FP interpolation ------------------------------
__global__ void three_nn(const float* __restrict__ xyz1, const float* __restrict__ xyz2,
                         int N1, int N2, int* __restrict__ nn_idx, float* __restrict__ nn_w)
{
    int n = blockIdx.x * blockDim.x + threadIdx.x;
    if (n >= N1) return;
    float px = xyz1[n * 3 + 0], py = xyz1[n * 3 + 1], pz = xyz1[n * 3 + 2];
    float d0 = 3.4e38f, d1 = 3.4e38f, d2 = 3.4e38f;
    int i0 = 0, i1 = 0, i2 = 0;
    for (int i = 0; i < N2; ++i) {
        float dx = xyz2[i * 3 + 0] - px;
        float dy = xyz2[i * 3 + 1] - py;
        float dz = xyz2[i * 3 + 2] - pz;
        float d = dx * dx + dy * dy + dz * dz;
        if (d < d0)      { d2 = d1; i2 = i1; d1 = d0; i1 = i0; d0 = d; i0 = i; }
        else if (d < d1) { d2 = d1; i2 = i1; d1 = d;  i1 = i; }
        else if (d < d2) { d2 = d;  i2 = i; }
    }
    float w0 = 1.0f / (d0 + 1e-8f);
    float w1 = 1.0f / (d1 + 1e-8f);
    float w2 = 1.0f / (d2 + 1e-8f);
    float ws = w0 + w1 + w2;
    nn_idx[n * 3 + 0] = i0; nn_idx[n * 3 + 1] = i1; nn_idx[n * 3 + 2] = i2;
    nn_w[n * 3 + 0] = w0 / ws; nn_w[n * 3 + 1] = w1 / ws; nn_w[n * 3 + 2] = w2 / ws;
}

__global__ void interp_gather(const float* __restrict__ f2, const int* __restrict__ nn_idx,
                              const float* __restrict__ nn_w, int N1, int C2,
                              float* __restrict__ out, int stride, int offset)
{
    long t = blockIdx.x * (long)blockDim.x + threadIdx.x;
    if (t >= (long)N1 * C2) return;
    int n = (int)(t / C2), c = (int)(t % C2);
    float v = 0.0f;
    #pragma unroll
    for (int k = 0; k < 3; ++k)
        v += f2[(long)nn_idx[n * 3 + k] * C2 + c] * nn_w[n * 3 + k];
    out[(long)n * stride + offset + c] = v;
}

__global__ void copy_cols(const float* __restrict__ src, int M, int Cs,
                          float* __restrict__ dst, int stride, int offset)
{
    long t = blockIdx.x * (long)blockDim.x + threadIdx.x;
    if (t >= (long)M * Cs) return;
    int r = (int)(t / Cs), c = (int)(t % Cs);
    dst[(long)r * stride + offset + c] = src[(long)r * Cs + c];
}

// ----------------------- image fusion helpers ------------------------------
__global__ void bilinear_gather(const float* __restrict__ img, const float* __restrict__ xy,
                                int N, int H, int W, float* __restrict__ out)
{
    int n = blockIdx.x * blockDim.x + threadIdx.x;
    if (n >= N) return;
    // xy normalized then denormalized in the reference => identity on pixel coords
    float x = xy[n * 2 + 0];
    float y = xy[n * 2 + 1];
    float x0 = floorf(x), y0 = floorf(y);
    float wx = x - x0, wy = y - y0;
    int x0i = min(max((int)x0, 0), W - 1);
    int x1i = min(x0i + 1, W - 1);
    int y0i = min(max((int)y0, 0), H - 1);
    int y1i = min(y0i + 1, H - 1);
    for (int c = 0; c < 3; ++c) {
        const float* f = img + (long)c * H * W;
        float v00 = f[y0i * W + x0i], v01 = f[y0i * W + x1i];
        float v10 = f[y1i * W + x0i], v11 = f[y1i * W + x1i];
        out[n * 3 + c] = v00 * (1 - wx) * (1 - wy) + v01 * wx * (1 - wy) +
                         v10 * (1 - wx) * wy + v11 * wx * wy;
    }
}

__global__ void attn_kernel(const float* __restrict__ pf, const float* __restrict__ imgf,
                            const float* __restrict__ fc1w, const float* __restrict__ fc1b,
                            const float* __restrict__ fc2w, const float* __restrict__ fc2b,
                            const float* __restrict__ fc3w, const float* __restrict__ fc3b,
                            int N, float* __restrict__ att)
{
    int n = blockIdx.x * blockDim.x + threadIdx.x;
    if (n >= N) return;
    float t[32];
    float i0 = imgf[n * 3 + 0], i1 = imgf[n * 3 + 1], i2 = imgf[n * 3 + 2];
    #pragma unroll
    for (int j = 0; j < 32; ++j)
        t[j] = fc1b[j] + fc2b[j] + i0 * fc1w[j] + i1 * fc1w[32 + j] + i2 * fc1w[64 + j];
    for (int c = 0; c < 128; ++c) {
        float p = pf[(long)n * 128 + c];
        #pragma unroll
        for (int j = 0; j < 32; ++j) t[j] += p * fc2w[c * 32 + j];
    }
    float s = fc3b[0];
    #pragma unroll
    for (int j = 0; j < 32; ++j) s += tanhf(t[j]) * fc3w[j];
    att[n] = 1.0f / (1.0f + expf(-s));
}

__global__ void fuse_concat(const float* __restrict__ pf, const float* __restrict__ img_new,
                            const float* __restrict__ att, int N, float* __restrict__ out)
{
    long t = blockIdx.x * (long)blockDim.x + threadIdx.x;
    if (t >= (long)N * 256) return;
    int n = (int)(t / 256), c = (int)(t % 256);
    out[t] = (c < 128) ? pf[(long)n * 128 + c]
                       : img_new[(long)n * 128 + (c - 128)] * att[n];
}

// ----------------------------- host side -----------------------------------
static const int NPTS[5]  = {16384, 4096, 1024, 256, 64};
static const int CFEAT[5] = {0, 96, 256, 512, 1024};
static const float RAD[4][2] = {{0.1f,0.5f},{0.5f,1.0f},{1.0f,2.0f},{2.0f,4.0f}};
static const int NS[4][2] = {{16,32},{16,32},{16,32},{16,32}};
static const int MLP[4][2][3] = {
    {{16,16,32},{32,32,64}}, {{64,64,128},{64,96,128}},
    {{128,196,256},{128,196,256}}, {{256,256,512},{256,384,512}}};
static const int FPC[4][2] = {{128,128},{256,256},{512,512},{512,512}};

static void run_mlp(const float* X, int M, int K, int Co,
                    const float* wgt, const float* g, const float* bb,
                    const float* bias, float* Y, float* stats, bool relu,
                    hipStream_t stream)
{
    dim3 grid((M + 63) / 64, (Co + 15) / 16);
    gemm_bias_wmma<<<grid, 128, 0, stream>>>(X, wgt, bias, Y, M, K, Co);
    colstats<<<Co, 256, 0, stream>>>(Y, M, Co, stats, stats + 1024);
    long tot = (long)M * Co;
    bn_act<<<(int)((tot + 255) / 256), 256, 0, stream>>>(Y, tot, Co,
        stats, stats + 1024, g, bb, relu ? 1 : 0);
}

extern "C" void kernel_launch(void* const* d_in, const int* in_sizes, int n_in,
                              void* d_out, int out_size, void* d_ws, size_t ws_size,
                              hipStream_t stream)
{
    (void)in_sizes; (void)n_in; (void)out_size; (void)ws_size;

    // jax pytree leaf order: sorted dict keys -> 'fp','fuse','head','sa'
    #define PAR(i) ((const float*)d_in[(i)])
    #define FP_Bb(k,l) PAR(3 + (k)*6 + (l)*3 + 0)
    #define FP_G(k,l)  PAR(3 + (k)*6 + (l)*3 + 1)
    #define FP_W(k,l)  PAR(3 + (k)*6 + (l)*3 + 2)
    #define FU(i) PAR(27 + (i))   // conv_b,conv_bb,conv_g,conv_w,fc1_b,fc1_w,fc2_b,fc2_w,fc3_b,fc3_w,fus_b,fus_bb,fus_g,fus_w
    #define HD(i) PAR(41 + (i))   // b1,b2,bb1,g1,w1,w2
    #define SA_Bb(lv,br,ly) PAR(47 + (lv)*18 + (br)*9 + (ly)*3 + 0)
    #define SA_G(lv,br,ly)  PAR(47 + (lv)*18 + (br)*9 + (ly)*3 + 1)
    #define SA_W(lv,br,ly)  PAR(47 + (lv)*18 + (br)*9 + (ly)*3 + 2)

    const float* pc  = (const float*)d_in[0];   // (16384,3)
    const float* img = (const float*)d_in[1];   // (3,384,1280)
    const float* xy  = (const float*)d_in[2];   // (16384,2)

    char* wp = (char*)d_ws;
    auto allocf = [&](size_t n) -> float* {
        float* p = (float*)wp; wp += ((n * 4) + 255) & ~(size_t)255; return p; };
    auto alloci = [&](size_t n) -> int* {
        int* p = (int*)wp; wp += ((n * 4) + 255) & ~(size_t)255; return p; };

    float* xyzL[5];
    xyzL[0] = const_cast<float*>(pc);             // never written
    for (int l = 1; l < 5; ++l) xyzL[l] = allocf((size_t)NPTS[l] * 3);
    float* featL[5]; featL[0] = nullptr;
    for (int l = 1; l < 5; ++l) featL[l] = allocf((size_t)NPTS[l] * CFEAT[l]);
    float* fpf[4];
    for (int i = 0; i < 4; ++i) fpf[i] = allocf((size_t)NPTS[i] * FPC[i][1]);
    int*   idxb  = alloci(4096);
    int*   gi    = alloci(4096 * 32);
    float* dists = allocf(16384);
    int*   nnI   = alloci(16384 * 3);
    float* nnW   = allocf(16384 * 3);
    float* stats = allocf(2048);
    float* img_g = allocf(16384 * 3);
    float* att   = allocf(16384);
    float* img_n = allocf((size_t)16384 * 128);
    float* Aar   = allocf((size_t)4608 * 1024);   // grouped / fp-concat inputs
    float* Y1    = allocf((size_t)8704 * 1024);   // gemm ping
    float* Y2    = allocf((size_t)8704 * 1024);   // gemm pong

    // ------------------------- SA (set abstraction) -------------------------
    for (int lvl = 0; lvl < 4; ++lvl) {
        int N = NPTS[lvl], S = NPTS[lvl + 1], Cf = CFEAT[lvl];
        const float* xin = xyzL[lvl];
        const float* fin = featL[lvl];
        fps_kernel<<<1, 1024, 0, stream>>>(xin, N, S, idxb, dists);
        gather_xyz<<<(S * 3 + 255) / 256, 256, 0, stream>>>(xin, idxb, S, xyzL[lvl + 1]);
        int coff = 0;
        for (int br = 0; br < 2; ++br) {
            int ns = NS[lvl][br];
            float r = RAD[lvl][br];
            ball_query_kernel<<<(S + 255) / 256, 256, 0, stream>>>(
                xyzL[lvl + 1], xin, S, N, r * r, ns, gi);
            int Ktot = 3 + Cf;
            long tot = (long)S * ns * Ktot;
            group_gather<<<(int)((tot + 255) / 256), 256, 0, stream>>>(
                xin, fin, xyzL[lvl + 1], gi, S, ns, Cf, Aar);
            int M = S * ns;
            const float* X = Aar; int Kc = Ktot;
            float* yb[2] = {Y1, Y2};
            for (int ly = 0; ly < 3; ++ly) {
                int Co = MLP[lvl][br][ly];
                float* Yd = yb[ly & 1];
                run_mlp(X, M, Kc, Co, SA_W(lvl, br, ly), SA_G(lvl, br, ly),
                        SA_Bb(lvl, br, ly), nullptr, Yd, stats, true, stream);
                X = Yd; Kc = Co;
            }
            maxpool<<<(int)(((long)S * Kc + 255) / 256), 256, 0, stream>>>(
                X, S, ns, Kc, featL[lvl + 1], CFEAT[lvl + 1], coff);
            coff += Kc;
        }
    }

    // ------------------------ FP (feature propagation) -----------------------
    const float* curF[5] = {nullptr, featL[1], featL[2], featL[3], featL[4]};
    int curC[5] = {0, 96, 256, 512, 1024};
    for (int i = 3; i >= 0; --i) {
        int N1 = NPTS[i], N2 = NPTS[i + 1];
        const float* f2 = curF[i + 1]; int C2 = curC[i + 1];
        const float* f1 = curF[i];     int C1 = curC[i];
        int Ktot = C2 + C1;
        three_nn<<<(N1 + 255) / 256, 256, 0, stream>>>(xyzL[i], xyzL[i + 1], N1, N2, nnI, nnW);
        interp_gather<<<(int)(((long)N1 * C2 + 255) / 256), 256, 0, stream>>>(
            f2, nnI, nnW, N1, C2, Aar, Ktot, 0);
        if (f1)
            copy_cols<<<(int)(((long)N1 * C1 + 255) / 256), 256, 0, stream>>>(
                f1, N1, C1, Aar, Ktot, C2);
        const float* X = Aar; int Kc = Ktot;
        for (int ly = 0; ly < 2; ++ly) {
            int Co = FPC[i][ly];
            float* Yd = (ly == 1) ? fpf[i] : Y1;
            run_mlp(X, N1, Kc, Co, FP_W(i, ly), FP_G(i, ly), FP_Bb(i, ly),
                    nullptr, Yd, stats, true, stream);
            X = Yd; Kc = Co;
        }
        curF[i] = fpf[i]; curC[i] = FPC[i][1];
    }

    // --------------------------- fusion + head ------------------------------
    const int N = 16384;
    bilinear_gather<<<(N + 255) / 256, 256, 0, stream>>>(img, xy, N, 384, 1280, img_g);
    attn_kernel<<<(N + 255) / 256, 256, 0, stream>>>(
        fpf[0], img_g, FU(5), FU(4), FU(7), FU(6), FU(9), FU(8), N, att);
    // img_new = relu(BN(img_g @ conv_w + conv_b))
    run_mlp(img_g, N, 3, 128, FU(3), FU(2), FU(1), FU(0), img_n, stats, true, stream);
    fuse_concat<<<(int)(((long)N * 256 + 255) / 256), 256, 0, stream>>>(
        fpf[0], img_n, att, N, Aar);
    // fused = relu(BN(concat @ fus_w + fus_b))
    run_mlp(Aar, N, 256, 128, FU(13), FU(12), FU(11), FU(10), Y1, stats, true, stream);
    // h = BN(fused @ w1 + b1)   (no relu)
    run_mlp(Y1, N, 128, 64, HD(4), HD(3), HD(2), HD(0), Y2, stats, false, stream);
    // out = h @ w2 + b2
    dim3 gfin((N + 63) / 64, 1);
    gemm_bias_wmma<<<gfin, 128, 0, stream>>>(Y2, HD(5), HD(1), (float*)d_out, N, 64, 3);
}
